// Wn0119EMsolo_38268158608164
// MI455X (gfx1250) — matrix operand order repaired and checked by reading, not compiled, not run on hardware
//
#include <hip/hip_runtime.h>
#include <math.h>

// ---------------- problem constants ----------------
#define NTT   2000
#define NSS   512
#define NHH   16
#define HID   256
static const long ROWS = (long)NTT * NSS;          // 1,024,000

// params layout (floats) inside workspace
#define P_KP 0
#define P_KS 16
#define P_KG 32
#define P_GP 48
#define P_GL 64
#define P_QB 80
#define P_RN 96        // 16x16 normalized taps, ga folded in
#define P_CP 352       // 16x3
#define P_CS 400
#define P_CG 448       // end 496

// workspace byte offsets
#define OFF_PARAMS 0
#define OFF_W1P    2048                         // 16*32*16 halfs = 16384B
#define OFF_W2P    (2048 + 16384)               // 3*8*32*16 halfs = 24576B
#define OFF_PS     65536                        // ROWS floats
#define OFF_PL     (65536 + ROWS*4)             // ROWS*16 floats
#define OFF_EV     (OFF_PL + ROWS*64)
#define OFF_VM     (OFF_EV + ROWS*64)

typedef __attribute__((ext_vector_type(16))) _Float16 v16h;
typedef __attribute__((ext_vector_type(8)))  _Float16 v8h;
typedef __attribute__((ext_vector_type(8)))  float    v8f;

__device__ __forceinline__ float sigm(float v) { return 1.f / (1.f + expf(-v)); }

__device__ __forceinline__ v16h frag16(const _Float16* p) {
    v8h lo = *(const v8h*)p;
    v8h hi = *(const v8h*)(p + 8);
    v16h r;
#pragma unroll
    for (int i = 0; i < 8; ++i) { r[i] = lo[i]; r[i + 8] = hi[i]; }
    return r;
}

// ---------------------------------------------------------------------------
// Kernel 1: precompute scalar params + pack W1/W2 into WMMA B fragments (f16)
// B fragment layout: lane = output column (lane&15), half i -> K = i + 16*(lane>=16)
// ---------------------------------------------------------------------------
__global__ __launch_bounds__(256)
void pack_params(const float* __restrict__ wR, const float* __restrict__ w,
                 const float* __restrict__ W1, const float* __restrict__ W2,
                 const float* __restrict__ cp, const float* __restrict__ cs,
                 const float* __restrict__ cg,
                 float* __restrict__ params, _Float16* __restrict__ w1p,
                 _Float16* __restrict__ w2p)
{
    int tid = threadIdx.x;
    if (tid < 16) {
        int h = tid;
        params[P_KP + h] = sigm(w[h]);
        params[P_KS + h] = sigm(w[16 + h]);
        params[P_KG + h] = sigm(w[32 + h]) * 0.1f;
        params[P_GP + h] = sigm(w[48 + h]);
        float gl = expf(w[64 + h]);
        params[P_GL + h] = gl * gl;
        params[P_QB + h] = fmaxf(w[80 + h], 0.f);
        // softmax over ga (16 values) — redundant per thread, trivial cost
        float mx = -1e30f;
        for (int j = 0; j < 16; ++j) mx = fmaxf(mx, w[96 + j]);
        float sum = 0.f;
        for (int j = 0; j < 16; ++j) sum += expf(w[96 + j] - mx);
        float ga = expf(w[96 + h] - mx) / sum;
        // normalized conv taps (relu(wR) per head), ga folded in
        float rsum = 0.f;
        for (int r = 0; r < 16; ++r) rsum += fmaxf(wR[h * 16 + r], 0.f);
        float inv = ga / (rsum + 1e-8f);
        for (int r = 0; r < 16; ++r)
            params[P_RN + h * 16 + r] = fmaxf(wR[h * 16 + r], 0.f) * inv;
        // cmap: relu(exp(c)-1), tiled row h -> c[h%4]
        for (int c = 0; c < 3; ++c) {
            params[P_CP + h * 3 + c] = fmaxf(expf(cp[(h & 3) * 3 + c]) - 1.f, 0.f);
            params[P_CS + h * 3 + c] = fmaxf(expf(cs[(h & 3) * 3 + c]) - 1.f, 0.f);
            params[P_CG + h * 3 + c] = fmaxf(expf(cg[(h & 3) * 3 + c]) - 1.f, 0.f);
        }
    }
    // W1: (6 x 256), pad K to 32. fragment idx = ((n*32+lane)*16 + i)
    for (int idx = tid; idx < 16 * 32 * 16; idx += blockDim.x) {
        int n = idx >> 9, rem = idx & 511, lane = rem >> 4, i = rem & 15;
        int K   = i + ((lane >= 16) ? 16 : 0);
        int col = n * 16 + (lane & 15);
        w1p[idx] = (K < 6) ? (_Float16)W1[K * HID + col] : (_Float16)0.f;
    }
    // W2: (256 x 48). fragment idx = (((nt*8+kc)*32+lane)*16 + i)
    for (int idx = tid; idx < 3 * 8 * 32 * 16; idx += blockDim.x) {
        int nt = idx >> 12, rem = idx & 4095, kc = rem >> 9, rem2 = rem & 511;
        int lane = rem2 >> 4, i = rem2 & 15;
        int K   = kc * 32 + i + ((lane >= 16) ? 16 : 0);
        int col = nt * 16 + (lane & 15);
        w2p[idx] = (_Float16)W2[K * 48 + col];
    }
}

// ---------------------------------------------------------------------------
// Kernel 2: MLP via WMMA f16 (v = tanh(x@W1+b1)@W2+b2) + per-row prep.
// 4 waves/block, 16 rows/wave. LDS transpose of the 16x256 hidden tile.
// Outputs: Ps[row], Pl[row][h], Ev[row][h], Vm[row][h]
// ---------------------------------------------------------------------------
__global__ __launch_bounds__(128)
void mlp_wmma(const float* __restrict__ x, const float* __restrict__ b1,
              const float* __restrict__ b2,
              const _Float16* __restrict__ w1p, const _Float16* __restrict__ w2p,
              float* __restrict__ Ps, float* __restrict__ Pl,
              float* __restrict__ Ev, float* __restrict__ Vm)
{
    __shared__ _Float16 Hsh[4][16][HID];     // 32 KB
    __shared__ float    rowScal[4][16][2];   // rP*P, E

    const int tid    = threadIdx.x;
    const int wave   = tid >> 5;
    const int lane   = tid & 31;
    const int laneLo = lane & 15;
    const int hi     = lane >> 4;
    const long rowBase = ((long)blockIdx.x * 4 + wave) * 16;

    // ---- A fragment for GEMM1: 16x32 f16, only K=0..5 nonzero (lanes 0-15) ----
    v16h a;
#pragma unroll
    for (int i = 0; i < 16; ++i) a[i] = (_Float16)0.f;

    if (hi == 0) {
        long row = rowBase + laneLo;
        float xr[6];
#pragma unroll
        for (int k = 0; k < 6; ++k) xr[k] = x[row * 6 + k];
#pragma unroll
        for (int k = 0; k < 6; ++k) a[k] = (_Float16)xr[k];
        float P = xr[0], E = xr[1], T1 = xr[2], T2 = xr[3];
        float frac = (T1 + T2) / (T2 - T1 + 1e-8f);
        frac = fminf(fmaxf(frac, -1.f), 1.f);
        float rP = 1.f - acosf(frac) * (1.f / 3.14159265358979323846f);
        rP = (T1 >= 0.f) ? 1.f : ((T2 <= 0.f) ? 0.f : rP);
        Ps[row] = (1.f - rP) * P;
        rowScal[wave][laneLo][0] = rP * P;
        rowScal[wave][laneLo][1] = E;
    }
    __syncthreads();

    // ---- GEMM1: 16 N-tiles of 16 -> hidden(256), tanh, transpose to LDS ----
#pragma unroll
    for (int n = 0; n < 16; ++n) {
        v16h b = frag16(w1p + (size_t)(n * 32 + lane) * 16);
        v8f c = {};
        c = __builtin_amdgcn_wmma_f32_16x16x32_f16(false, a, false, b,
                                                   (short)0, c, false, false);
        float bias = b1[n * 16 + laneLo];
#pragma unroll
        for (int r = 0; r < 8; ++r) {
            int m = r + 8 * hi;                       // C layout: M = r + 8*hi
            Hsh[wave][m][n * 16 + laneLo] = (_Float16)tanhf(c[r] + bias);
        }
    }
    __syncthreads();

    // ---- GEMM2: K=256 (8 chunks of 32), N=48 (3 tiles) ----
    v8f acc0 = {}, acc1 = {}, acc2 = {};
#pragma unroll
    for (int kc = 0; kc < 8; ++kc) {
        // A layout: lane=M(=lane&15); half i -> K=kc*32+16*hi+i
        int base = kc * 32 + 16 * hi;
        v8h lo  = *(const v8h*)&Hsh[wave][laneLo][base];
        v8h hi8 = *(const v8h*)&Hsh[wave][laneLo][base + 8];
        v16h a2;
#pragma unroll
        for (int i = 0; i < 8; ++i) { a2[i] = lo[i]; a2[i + 8] = hi8[i]; }

        v16h b0 = frag16(w2p + (size_t)((0 * 8 + kc) * 32 + lane) * 16);
        acc0 = __builtin_amdgcn_wmma_f32_16x16x32_f16(false, a2, false, b0,
                                                      (short)0, acc0, false, false);
        v16h b1f = frag16(w2p + (size_t)((1 * 8 + kc) * 32 + lane) * 16);
        acc1 = __builtin_amdgcn_wmma_f32_16x16x32_f16(false, a2, false, b1f,
                                                      (short)0, acc1, false, false);
        v16h b2f = frag16(w2p + (size_t)((2 * 8 + kc) * 32 + lane) * 16);
        acc2 = __builtin_amdgcn_wmma_f32_16x16x32_f16(false, a2, false, b2f,
                                                      (short)0, acc2, false, false);
    }

    // ---- epilogue: activations + per-row scalars -> Pl / Ev / Vm ----
    float bvi = b2[laneLo], bve = b2[16 + laneLo], bvm = b2[32 + laneLo];
#pragma unroll
    for (int r = 0; r < 8; ++r) {
        int m = r + 8 * hi;
        long row = rowBase + m;
        float rpP = rowScal[wave][m][0];
        float Em  = rowScal[wave][m][1];
        float vi  = acc0[r] + bvi;                      // hard_sigmoid(2*vi)
        vi = fminf(fmaxf(vi * (1.f / 3.f) + 0.5f, 0.f), 1.f);
        float ve  = fmaxf(acc1[r] + bve, 0.f) * 2.f;
        float vmv = expf(acc2[r] + bvm);
        size_t o = (size_t)row * NHH + laneLo;
        Pl[o] = rpP * vi;
        Ev[o] = Em * ve;
        Vm[o] = vmv;
    }
}

// ---------------------------------------------------------------------------
// Kernel 3: sequential scan over t, fused causal conv (16 taps, ga folded),
// head reductions via wave32 shuffles, final output (NT,NS,4).
// One thread per (s,h); 16-lane half-wave == one site's 16 heads.
// ---------------------------------------------------------------------------
__global__ __launch_bounds__(256)
void scan_conv_out(const float* __restrict__ params,
                   const float* __restrict__ Ps, const float* __restrict__ Pl,
                   const float* __restrict__ Ev, const float* __restrict__ Vm,
                   float* __restrict__ out)
{
    int gid = blockIdx.x * blockDim.x + threadIdx.x;   // 0 .. NS*NH-1
    int s = gid >> 4;
    int h = gid & 15;

    float kp = params[P_KP + h], ks = params[P_KS + h], kg = params[P_KG + h];
    float gp = params[P_GP + h], gL = params[P_GL + h], qb = params[P_QB + h];
    float rn[16];
#pragma unroll
    for (int r = 0; r < 16; ++r) rn[r] = params[P_RN + h * 16 + r];
    float cpm[3], csm[3], cgm[3];
#pragma unroll
    for (int c = 0; c < 3; ++c) {
        cpm[c] = params[P_CP + h * 3 + c];
        csm[c] = params[P_CS + h * 3 + c];
        cgm[c] = params[P_CG + h * 3 + c];
    }

    float bp[16], bs[16], bg[16];
#pragma unroll
    for (int r = 0; r < 16; ++r) { bp[r] = 0.f; bs[r] = 0.f; bg[r] = 0.f; }
    float Sf = 0.f, Ss = 0.f, Sg = 0.f;

    for (int t = 0; t < NTT; ++t) {
        size_t rowi = (size_t)t * NSS + s;
        size_t o = rowi * NHH + h;
        if (t + 8 < NTT) {                      // -> global_prefetch_b8
            __builtin_prefetch(&Pl[o + (size_t)8 * NSS * NHH], 0, 1);
        }
        float Pst = Ps[rowi];
        float Plt = Pl[o], Evt = Ev[o], vmt = Vm[o];

        float Sf1 = fmaxf(Sf + Pst, 0.f);
        float qf  = fminf(Sf1, vmt);
        Sf = Sf1 - qf;
        float H  = fmaxf(Ss + Plt + qf - Evt, 0.f);
        float qp = fmaxf(kp * (H - gL), 0.f);
        float qs = ks * fminf(H, gL);
        Ss = H - qp - qs;
        float Sg1 = Sg + qs * gp;
        float qg  = Sg1 * kg + qb;
        Sg = fmaxf(Sg1 - qg, 0.f);
        float qsO = qs * (1.f - gp);

        // shift register ring buffers (static indices -> stays in VGPRs)
#pragma unroll
        for (int r = 15; r > 0; --r) { bp[r] = bp[r - 1]; bs[r] = bs[r - 1]; bg[r] = bg[r - 1]; }
        bp[0] = qp; bs[0] = qsO; bg[0] = qg;

        float cP = 0.f, cS = 0.f, cG = 0.f;
#pragma unroll
        for (int r = 0; r < 16; ++r) {
            cP = fmaf(rn[r], bp[r], cP);
            cS = fmaf(rn[r], bs[r], cS);
            cG = fmaf(rn[r], bg[r], cG);
        }
        // rn already includes ga: cP/cS/cG == QpR/QsR/QgR for this head
        float qsum = cP + cS + cG;
        float n0 = cP * cpm[0] + cS * csm[0] + cG * cgm[0];
        float n1 = cP * cpm[1] + cS * csm[1] + cG * cgm[1];
        float n2 = cP * cpm[2] + cS * csm[2] + cG * cgm[2];
        // butterfly reduce over the 16 heads (stays inside half-wave, wave32)
#pragma unroll
        for (int m = 1; m < 16; m <<= 1) {
            qsum += __shfl_xor(qsum, m, 32);
            n0   += __shfl_xor(n0, m, 32);
            n1   += __shfl_xor(n1, m, 32);
            n2   += __shfl_xor(n2, m, 32);
        }
        if (h == 0) {
            float Qd = qsum + 1e-8f;
            float4 o4 = make_float4(qsum, n0 / Qd, n1 / Qd, n2 / Qd);
            ((float4*)out)[rowi] = o4;
        }
    }
}

// ---------------------------------------------------------------------------
extern "C" void kernel_launch(void* const* d_in, const int* in_sizes, int n_in,
                              void* d_out, int out_size, void* d_ws, size_t ws_size,
                              hipStream_t stream)
{
    const float* x  = (const float*)d_in[0];
    const float* wR = (const float*)d_in[1];
    const float* w  = (const float*)d_in[2];
    const float* W1 = (const float*)d_in[3];
    const float* b1 = (const float*)d_in[4];
    const float* W2 = (const float*)d_in[5];
    const float* b2 = (const float*)d_in[6];
    const float* cp = (const float*)d_in[7];
    const float* cs = (const float*)d_in[8];
    const float* cg = (const float*)d_in[9];

    char* ws = (char*)d_ws;
    float*    params = (float*)(ws + OFF_PARAMS);
    _Float16* w1p    = (_Float16*)(ws + OFF_W1P);
    _Float16* w2p    = (_Float16*)(ws + OFF_W2P);
    float*    Ps     = (float*)(ws + OFF_PS);
    float*    Pl     = (float*)(ws + OFF_PL);
    float*    Ev     = (float*)(ws + OFF_EV);
    float*    Vm     = (float*)(ws + OFF_VM);

    pack_params<<<1, 256, 0, stream>>>(wR, w, W1, W2, cp, cs, cg, params, w1p, w2p);

    int mlpBlocks = (int)(ROWS / 64);           // 4 waves x 16 rows per block
    mlp_wmma<<<mlpBlocks, 128, 0, stream>>>(x, b1, b2, w1p, w2p, Ps, Pl, Ev, Vm);

    int scanBlocks = (NSS * NHH) / 256;         // 32 blocks
    scan_conv_out<<<scanBlocks, 256, 0, stream>>>(params, Ps, Pl, Ev, Vm, (float*)d_out);
}
